// FirstStageDecoderBlock_27754078667056
// MI455X (gfx1250) — matrix operand order repaired and checked by reading, not compiled, not run on hardware
//
#include <hip/hip_runtime.h>
#include <hip/hip_bf16.h>

typedef __attribute__((ext_vector_type(16))) _Float16 v16h;
typedef __attribute__((ext_vector_type(8)))  _Float16 v8h;
typedef __attribute__((ext_vector_type(8)))  float    v8f;
typedef __attribute__((ext_vector_type(4)))  int      v4i;

#define DEV_LEAKY(v) ((v) >= 0.f ? (v) : 0.2f * (v))

// Optional gfx1250 async global->LDS copy path (guarded; falls back to sync loads)
#if defined(__has_builtin)
#if __has_builtin(__builtin_amdgcn_global_load_async_to_lds_b128) && \
    __has_builtin(__builtin_amdgcn_s_wait_asynccnt)
#define HAVE_ASYNC_LDS 1
#endif
#endif
#ifndef HAVE_ASYNC_LDS
#define HAVE_ASYNC_LDS 0
#endif

#define AS_GLOBAL __attribute__((address_space(1)))
#define AS_LDS    __attribute__((address_space(3)))

// ---------------- problem constants ----------------
static constexpr int BB = 4, TT = 10, CC = 96, HH = 32, WW = 32, LL = 13, OUTC = 96;
static constexpr int HWPIX = HH * WW;                 // 1024
static constexpr long N_ALL = (long)TT * BB * HWPIX;  // 40960
static constexpr long N_STEP = (long)BB * HWPIX;      // 4096

// ---------------- generic implicit-GEMM WMMA kernel ----------------
// C[M x N] = A[M x K] * im2col(src)[K x N]  (+bias, +add, act, store modes)
// REQUIREMENTS (hold for every call here): K % 32 == 0; if aIsF16 then M % 64 == 0.
struct GemmArgs {
  const void* A; int aIsF16; int M, N, K;
  const float* bias;          // nullable, indexed [gm]
  const float* addSrc;        // nullable, indexed [gm*addLd + gn]
  long addLd;
  int act;                    // 0 none, 1 leaky
  int storeMode;              // 0 dense [gm*ldc+gn], 1 conv-transpose phase scatter
  float* outDense; long ldc;
  float* outScatter; int py, px;
  // im2col source description: value(b,t,cin,iy,ix) with column n -> (img,y,x),
  // img = t*4+b (strideT==0 collapses to img=b for 4-image tensors)
  const float* src; long strideB, strideT, chStride;
  int Hs, Ws, Cin, KH, KW, padY, padX;
};

template <int AISF16>
__launch_bounds__(256)
__global__ void wmma_conv_gemm(GemmArgs g) {
  alignas(16) __shared__ _Float16 As[64 * 32];  // [m][k]
  alignas(16) __shared__ _Float16 Bs[64 * 32];  // [n][k]

  const int tid  = threadIdx.x;
  const int wave = tid >> 5;
  const int lane = tid & 31;
  const int mt = wave & 3;           // 0..3 -> 64 rows
  const int nt = wave >> 2;          // 0..1 -> two 32-col halves
  const int mBase = blockIdx.y * 64;
  const int nBase = blockIdx.x * 64;
  const int KHW = g.KH * g.KW;
  const int HW  = g.Hs * g.Ws;
  const int Hs = g.Hs, Ws = g.Ws, padY = g.padY, padX = g.padX, KW = g.KW;

  // ---- B gather coordinates: thread -> (n = tid>>2, k = (tid&3)*8 .. +7) ----
  const int nLocal = tid >> 2;          // 0..63
  const int kColB  = (tid & 3) * 8;     // 0,8,16,24
  const int gnLd = nBase + nLocal;
  const bool nOk = gnLd < g.N;
  const int nSafe = nOk ? gnLd : 0;
  const int img = nSafe / HW, pPix = nSafe % HW;
  const int yPix = pPix / Ws, xPix = pPix % Ws;
  const float* __restrict__ srcBT =
      g.src + (long)(img & 3) * g.strideB + (long)(img >> 2) * g.strideT;

  // ---- A load coordinates: thread -> (m = tid>>2, k = (tid&3)*8 .. +7) ----
  const int aRow = tid >> 2;            // 0..63
  const int aCol = (tid & 3) * 8;       // 0,8,16,24
  const int gmA = mBase + aRow;
  const bool aOk = gmA < g.M;
  const int gmAc = min(gmA, g.M - 1);   // clamped row: loads always in-bounds

  v8f acc[2] = {v8f{}, v8f{}};

  for (int k0 = 0; k0 < g.K; k0 += 32) {
    // ---------------- stage A tile ----------------
    if (AISF16) {
#if HAVE_ASYNC_LDS
      // async DMA: 16B per lane, global -> LDS (M % 64 == 0 guaranteed for f16 A)
      const _Float16* gp = (const _Float16*)g.A + (long)gmA * g.K + k0 + aCol;
      __builtin_amdgcn_global_load_async_to_lds_b128(
          (AS_GLOBAL v4i*)gp,
          (AS_LDS v4i*)&As[aRow * 32 + aCol], 0, 0);
#else
      const long off = (long)gmAc * g.K + k0 + aCol;
      v8h av = *reinterpret_cast<const v8h*>((const _Float16*)g.A + off);
#pragma unroll
      for (int e = 0; e < 8; ++e) av[e] = aOk ? av[e] : (_Float16)0.f;
      if (k0 + 32 < g.K) __builtin_prefetch((const _Float16*)g.A + off + 32, 0, 1);
      *reinterpret_cast<v8h*>(&As[aRow * 32 + aCol]) = av;
#endif
    } else {
      const long off = (long)gmAc * g.K + k0 + aCol;
      const float4 f0 = *reinterpret_cast<const float4*>((const float*)g.A + off);
      const float4 f1 = *reinterpret_cast<const float4*>((const float*)g.A + off + 4);
      v8h av;
      av[0] = (_Float16)f0.x; av[1] = (_Float16)f0.y;
      av[2] = (_Float16)f0.z; av[3] = (_Float16)f0.w;
      av[4] = (_Float16)f1.x; av[5] = (_Float16)f1.y;
      av[6] = (_Float16)f1.z; av[7] = (_Float16)f1.w;
#pragma unroll
      for (int e = 0; e < 8; ++e) av[e] = aOk ? av[e] : (_Float16)0.f;
      if (k0 + 32 < g.K) __builtin_prefetch((const float*)g.A + off + 32, 0, 1);
      *reinterpret_cast<v8h*>(&As[aRow * 32 + aCol]) = av;
    }

    // ---------------- stage B tile: batched branch-free im2col gather ----------------
    // phase 1: addresses + masks, phase 2: 8 independent loads, phase 3: convert/select
    long addrv[8];
    bool okm[8];
#pragma unroll
    for (int j = 0; j < 8; ++j) {
      const int gk = k0 + kColB + j;          // gk < K always (K % 32 == 0)
      const int cin = gk / KHW, r = gk % KHW;
      const int ky = r / KW, kx = r % KW;
      const int iy = yPix + ky - padY;
      const int ix = xPix + kx - padX;
      okm[j] = nOk & (iy >= 0) & (iy < Hs) & (ix >= 0) & (ix < Ws);
      const int iyc = min(max(iy, 0), Hs - 1);
      const int ixc = min(max(ix, 0), Ws - 1);
      addrv[j] = (long)cin * g.chStride + iyc * Ws + ixc;
    }
    float vals[8];
#pragma unroll
    for (int j = 0; j < 8; ++j) vals[j] = srcBT[addrv[j]];
    v8h bv;
#pragma unroll
    for (int j = 0; j < 8; ++j) bv[j] = (_Float16)(okm[j] ? vals[j] : 0.f);
    *reinterpret_cast<v8h*>(&Bs[nLocal * 32 + kColB]) = bv;

#if HAVE_ASYNC_LDS
    if (AISF16) __builtin_amdgcn_s_wait_asynccnt(0);
#endif
    __syncthreads();

    // ---- fragments per ISA 7.12.2; one A frag feeds two WMMAs ----
    const int mRow = mt * 16 + (lane & 15);
    const int aC0  = (lane < 16) ? 0 : 8;     // K chunks {0..7,16..23} / {8..15,24..31}
    const v8h a0 = *reinterpret_cast<const v8h*>(&As[mRow * 32 + aC0]);
    const v8h a1 = *reinterpret_cast<const v8h*>(&As[mRow * 32 + aC0 + 16]);
    const v16h afrag = __builtin_shufflevector(a0, a1, 0,1,2,3,4,5,6,7,8,9,10,11,12,13,14,15);

    const int bC0 = (lane < 16) ? 0 : 8;      // K chunks {0..7,16..23} / {8..15,24..31}
    const int nCol0 = nt * 32 + (lane & 15);
    const v8h b0 = *reinterpret_cast<const v8h*>(&Bs[nCol0 * 32 + bC0]);
    const v8h b1 = *reinterpret_cast<const v8h*>(&Bs[nCol0 * 32 + bC0 + 16]);
    const v16h bf0 = __builtin_shufflevector(b0, b1, 0,1,2,3,4,5,6,7,8,9,10,11,12,13,14,15);
    const int nCol1 = nCol0 + 16;
    const v8h b2 = *reinterpret_cast<const v8h*>(&Bs[nCol1 * 32 + bC0]);
    const v8h b3 = *reinterpret_cast<const v8h*>(&Bs[nCol1 * 32 + bC0 + 16]);
    const v16h bf1 = __builtin_shufflevector(b2, b3, 0,1,2,3,4,5,6,7,8,9,10,11,12,13,14,15);

    acc[0] = __builtin_amdgcn_wmma_f32_16x16x32_f16(false, afrag, false, bf0,
                                                    (short)0, acc[0], false, false);
    acc[1] = __builtin_amdgcn_wmma_f32_16x16x32_f16(false, afrag, false, bf1,
                                                    (short)0, acc[1], false, false);
    __syncthreads();
  }

  // ---- epilogue: C layout -> lanes 0-15: N=lane,M=r ; lanes 16-31: N=lane-16,M=8+r ----
  const int mOff = mBase + mt * 16 + ((lane < 16) ? 0 : 8);
#pragma unroll
  for (int s = 0; s < 2; ++s) {
    const int gn = nBase + nt * 32 + s * 16 + (lane & 15);
    if (gn < g.N) {
#pragma unroll
      for (int r = 0; r < 8; ++r) {
        const int gm = mOff + r;
        if (gm < g.M) {
          float v = acc[s][r];
          if (g.bias)   v += g.bias[gm];
          if (g.addSrc) v += g.addSrc[(long)gm * g.addLd + gn];
          if (g.act == 1) v = DEV_LEAKY(v);
          if (g.storeMode == 0) {
            g.outDense[(long)gm * g.ldc + gn] = v;
          } else {
            const int b = gn >> 10, p = gn & 1023, oy = p >> 5, ox = p & 31;
            const long o = (((long)b * 960 + gm) * 64 + (2 * oy + g.py)) * 64 + (2 * ox + g.px);
            g.outScatter[o] = v;
          }
        }
      }
    }
  }
}

// ---------------- bilinear trajectory warp: warped[b][l*C+c][p] ----------------
__global__ void warp_kernel(const float* __restrict__ h,      // (B,C,32,32)
                            const float* __restrict__ flow,   // [2L][4096]
                            float* __restrict__ warped) {     // (B,L*C,32,32)
  const int idx = blockIdx.x * blockDim.x + threadIdx.x;
  if (idx >= BB * LL * HWPIX) return;
  const int p = idx & 1023;
  const int bl = idx >> 10;
  const int l = bl % LL, b = bl / LL;
  const int y = p >> 5, x = p & 31;
  const long n = (long)b * 1024 + p;
  const float fx = flow[(long)(2 * l) * N_STEP + n];
  const float fy = flow[(long)(2 * l + 1) * N_STEP + n];
  const float gx = (float)x - fx, gy = (float)y - fy;
  const float x0f = floorf(gx), y0f = floorf(gy);
  const int x0 = (int)x0f, y0 = (int)y0f;
  const float wx = gx - x0f, wy = gy - y0f;
  const int x1 = x0 + 1, y1 = y0 + 1;
  const bool vx0 = (x0 >= 0) && (x0 <= 31), vx1 = (x1 >= 0) && (x1 <= 31);
  const bool vy0 = (y0 >= 0) && (y0 <= 31), vy1 = (y1 >= 0) && (y1 <= 31);
  const int xc0 = min(max(x0, 0), 31), xc1 = min(max(x1, 0), 31);
  const int yc0 = min(max(y0, 0), 31), yc1 = min(max(y1, 0), 31);
  const int o00 = yc0 * 32 + xc0, o01 = yc0 * 32 + xc1;
  const int o10 = yc1 * 32 + xc0, o11 = yc1 * 32 + xc1;
  const float w00 = ((vx0 && vy0) ? 1.f : 0.f) * (1.f - wx) * (1.f - wy);
  const float w01 = ((vx1 && vy0) ? 1.f : 0.f) * wx * (1.f - wy);
  const float w10 = ((vx0 && vy1) ? 1.f : 0.f) * (1.f - wx) * wy;
  const float w11 = ((vx1 && vy1) ? 1.f : 0.f) * wx * wy;
  const float* hb = h + (long)b * CC * HWPIX;
  float* ob = warped + ((long)b * (LL * CC) + (long)l * CC) * HWPIX + p;
  for (int c = 0; c < CC; ++c) {
    const float* hc = hb + (long)c * HWPIX;
    ob[(long)c * HWPIX] = hc[o00] * w00 + hc[o01] * w01 + hc[o10] * w10 + hc[o11] * w11;
  }
}

// ---------------- GRU gating ----------------
__global__ void gating_kernel(const float* __restrict__ i2h,   // [3C][40960]
                              const float* __restrict__ h2h,   // [3C][4096]
                              const float* __restrict__ hprev, // (B,C,32,32)
                              float* __restrict__ hnext,       // (B,C,32,32)
                              float* __restrict__ yseq,        // (B,T*C,32,32)
                              int t) {
  const int idx = blockIdx.x * blockDim.x + threadIdx.x;
  if (idx >= BB * CC * HWPIX) return;
  const int p = idx & 1023;
  const int bc = idx >> 10;
  const int c = bc % CC, b = bc / CC;
  const long n = (long)b * 1024 + p;
  const long col = (long)t * N_STEP + n;
  const float ir = i2h[(long)c * N_ALL + col];
  const float iu = i2h[(long)(CC + c) * N_ALL + col];
  const float im = i2h[(long)(2 * CC + c) * N_ALL + col];
  const float hr = h2h[(long)c * N_STEP + n];
  const float hu = h2h[(long)(CC + c) * N_STEP + n];
  const float hm = h2h[(long)(2 * CC + c) * N_STEP + n];
  const float rg = 1.f / (1.f + expf(-(ir + hr)));
  const float ug = 1.f / (1.f + expf(-(iu + hu)));
  float nm = im + rg * hm;
  nm = DEV_LEAKY(nm);
  const float hp = hprev[(long)(b * CC + c) * HWPIX + p];
  const float hn = ug * hp + (1.f - ug) * nm;
  hnext[(long)(b * CC + c) * HWPIX + p] = hn;
  yseq[((long)b * (TT * CC) + (long)t * CC + c) * HWPIX + p] = hn;
}

// ---------------- pack ConvTranspose weights into 4 phase GEMM matrices (f16) ----------------
// Wct[ph][oc][ic*4 + ty*2 + tx] = ct_w[ic][oc][3 - (2*ty+py)][3 - (2*tx+px)]
__global__ void pack_ct_kernel(const float* __restrict__ ct_w, _Float16* __restrict__ wct) {
  const long idx = (long)blockIdx.x * blockDim.x + threadIdx.x;
  const long total = 4L * 960 * 960 * 4;
  if (idx >= total) return;
  const int tx = (int)(idx & 1);
  const int ty = (int)((idx >> 1) & 1);
  long r = idx >> 2;
  const int ic = (int)(r % 960); r /= 960;
  const int oc = (int)(r % 960); r /= 960;
  const int ph = (int)r;                 // ph = py*2 + px
  const int py = ph >> 1, px = ph & 1;
  const float v = ct_w[(((long)ic * 960 + oc) * 4 + (3 - (2 * ty + py))) * 4 + (3 - (2 * tx + px))];
  wct[((long)ph * 960 + oc) * 3840 + (long)ic * 4 + ty * 2 + tx] = (_Float16)v;
}

// ---------------- host side ----------------
static inline void launch_gemm(const GemmArgs& g, hipStream_t s) {
  dim3 grid((g.N + 63) / 64, (g.M + 63) / 64);
  if (g.aIsF16) wmma_conv_gemm<1><<<grid, dim3(256), 0, s>>>(g);
  else          wmma_conv_gemm<0><<<grid, dim3(256), 0, s>>>(g);
}

extern "C" void kernel_launch(void* const* d_in, const int* in_sizes, int n_in,
                              void* d_out, int out_size, void* d_ws, size_t ws_size,
                              hipStream_t stream) {
  const float* x      = (const float*)d_in[0];
  const float* h0     = (const float*)d_in[1];
  const float* i2h_w  = (const float*)d_in[2];
  const float* i2h_b  = (const float*)d_in[3];
  const float* i2f_w  = (const float*)d_in[4];
  const float* i2f_b  = (const float*)d_in[5];
  const float* h2f_w  = (const float*)d_in[6];
  const float* h2f_b  = (const float*)d_in[7];
  const float* flow_w = (const float*)d_in[8];
  const float* flow_b = (const float*)d_in[9];
  const float* ret_w  = (const float*)d_in[10];
  const float* ret_b  = (const float*)d_in[11];
  const float* ct_w   = (const float*)d_in[12];
  const float* ct_b   = (const float*)d_in[13];
  float* out = (float*)d_out;

  // workspace layout (floats)
  float* ws = (float*)d_ws;
  const long sz_i2h  = 3L * CC * N_ALL;        // 11,796,480
  const long sz_i2f  = 32L * N_ALL;            //  1,310,720
  const long sz_f    = 32L * N_STEP;
  const long sz_flow = 2L * LL * N_STEP;
  const long sz_warp = (long)BB * LL * CC * HWPIX;
  const long sz_h2h  = 3L * CC * N_STEP;
  const long sz_h    = (long)BB * CC * HWPIX;
  const long sz_yseq = (long)BB * TT * CC * HWPIX;
  float* ws_i2h  = ws;                long o = sz_i2h;
  float* ws_i2f  = ws + o;            o += sz_i2f;
  float* ws_f    = ws + o;            o += sz_f;
  float* ws_flow = ws + o;            o += sz_flow;
  float* ws_warp = ws + o;            o += sz_warp;
  float* ws_h2h  = ws + o;            o += sz_h2h;
  float* ws_hA   = ws + o;            o += sz_h;
  float* ws_hB   = ws + o;            o += sz_h;
  float* ws_yseq = ws + o;            o += sz_yseq;
  _Float16* ws_wct = (_Float16*)(ws + o);      // 4*960*3840 halves

  // pack ConvTranspose weights (independent; runs up front)
  {
    const long total = 4L * 960 * 960 * 4;
    pack_ct_kernel<<<dim3((unsigned)((total + 255) / 256)), dim3(256), 0, stream>>>(ct_w, ws_wct);
  }

  GemmArgs g{};
  // ---- i2h over all T*B frames: 3x3, pad 1 ----
  g.A = i2h_w; g.aIsF16 = 0; g.M = 3 * CC; g.N = (int)N_ALL; g.K = CC * 9;
  g.bias = i2h_b; g.addSrc = nullptr; g.addLd = 0; g.act = 0;
  g.storeMode = 0; g.outDense = ws_i2h; g.ldc = N_ALL; g.outScatter = nullptr; g.py = g.px = 0;
  g.src = x; g.strideB = (long)TT * CC * HWPIX; g.strideT = (long)CC * HWPIX; g.chStride = HWPIX;
  g.Hs = HH; g.Ws = WW; g.Cin = CC; g.KH = 3; g.KW = 3; g.padY = 1; g.padX = 1;
  launch_gemm(g, stream);

  // ---- i2f over all frames: 5x5, pad 2 ----
  g.A = i2f_w; g.M = 32; g.K = CC * 25; g.bias = i2f_b;
  g.outDense = ws_i2f; g.KH = 5; g.KW = 5; g.padY = 2; g.padX = 2;
  launch_gemm(g, stream);

  // ---- recurrence ----
  const float* hcur = h0;
  for (int t = 0; t < TT; ++t) {
    float* hnxt = (t & 1) ? ws_hB : ws_hA;

    // h2f: 5x5 pad2 over h, fused (+ i2f_t, leaky)
    g.A = h2f_w; g.aIsF16 = 0; g.M = 32; g.N = (int)N_STEP; g.K = CC * 25;
    g.bias = h2f_b; g.addSrc = ws_i2f + (long)t * N_STEP; g.addLd = N_ALL; g.act = 1;
    g.storeMode = 0; g.outDense = ws_f; g.ldc = N_STEP;
    g.src = hcur; g.strideB = (long)CC * HWPIX; g.strideT = 0; g.chStride = HWPIX;
    g.Cin = CC; g.KH = 5; g.KW = 5; g.padY = 2; g.padX = 2;
    launch_gemm(g, stream);

    // flow: 5x5 pad2 over f (f stored channel-major [32][4096])
    g.A = flow_w; g.M = 2 * LL; g.K = 32 * 25; g.bias = flow_b; g.addSrc = nullptr; g.act = 0;
    g.outDense = ws_flow;
    g.src = ws_f; g.strideB = HWPIX; g.strideT = 0; g.chStride = N_STEP;
    g.Cin = 32; g.KH = 5; g.KW = 5; g.padY = 2; g.padX = 2;
    launch_gemm(g, stream);

    // bilinear trajectory warp
    warp_kernel<<<dim3((BB * LL * HWPIX + 255) / 256), dim3(256), 0, stream>>>(hcur, ws_flow, ws_warp);

    // ret: 1x1 over warped (K = L*C = 1248)
    g.A = ret_w; g.M = 3 * CC; g.K = LL * CC; g.bias = ret_b;
    g.outDense = ws_h2h;
    g.src = ws_warp; g.strideB = (long)LL * CC * HWPIX; g.strideT = 0; g.chStride = HWPIX;
    g.Cin = LL * CC; g.KH = 1; g.KW = 1; g.padY = 0; g.padX = 0;
    launch_gemm(g, stream);

    // GRU gating -> h_next and y_seq
    gating_kernel<<<dim3((BB * CC * HWPIX + 255) / 256), dim3(256), 0, stream>>>(
        ws_i2h, ws_h2h, hcur, hnxt, ws_yseq, t);

    hcur = hnxt;
  }

  // ---- ConvTranspose2d as 4 phase GEMMs (M=960, N=4096, K=3840) ----
  for (int ph = 0; ph < 4; ++ph) {
    const int py = ph >> 1, px = ph & 1;
    g.A = ws_wct + (long)ph * 960 * 3840; g.aIsF16 = 1;
    g.M = 960; g.N = (int)N_STEP; g.K = 3840;
    g.bias = ct_b; g.addSrc = nullptr; g.act = 1;
    g.storeMode = 1; g.outDense = nullptr; g.ldc = 0;
    g.outScatter = out; g.py = py; g.px = px;
    g.src = ws_yseq; g.strideB = (long)TT * CC * HWPIX; g.strideT = 0; g.chStride = HWPIX;
    g.Cin = TT * CC; g.KH = 2; g.KW = 2; g.padY = 1 - py; g.padX = 1 - px;
    launch_gemm(g, stream);
  }
}
